// Conv2d_59777354825943
// MI455X (gfx1250) — matrix operand order repaired and checked by reading, not compiled, not run on hardware
//
#include <hip/hip_runtime.h>

typedef __bf16 bf16;
typedef __attribute__((ext_vector_type(8)))  __bf16 v8bf;
typedef __attribute__((ext_vector_type(16))) __bf16 v16bf;
typedef __attribute__((ext_vector_type(8)))  float  v8f;
typedef unsigned int u32x4 __attribute__((ext_vector_type(4)));
typedef int          i32x4 __attribute__((ext_vector_type(4)));
typedef int          i32x8 __attribute__((ext_vector_type(8)));

// LDS (addrspace 3) views so fragment reads select ds_load_b128, not flat.
typedef __attribute__((address_space(3))) const bf16 lds_cbf16;
typedef __attribute__((address_space(3))) const v8bf lds_cv8bf;

#define C_IN    64
#define C_OUT   128
#define HW      112
#define KTAPS   9
#define NT      4      // output h-rows per block (N tile = NT*16 = 64 pixels)
#define LDS_STR 88     // padded LDS row stride (elems): 176B rows -> conflict-free

// ---------------------------------------------------------------------------
// Pre-pack weights: (O=128, C=64, 9) f32, k = c*9+tap  ->  bf16 Wp[tap][o][c]
// ---------------------------------------------------------------------------
__global__ __launch_bounds__(256) void pack_weights(const float* __restrict__ w,
                                                    bf16* __restrict__ wp) {
    int idx = blockIdx.x * blockDim.x + threadIdx.x;      // 9*128*64 = 73728
    int tap = idx / (C_OUT * C_IN);
    int rem = idx % (C_OUT * C_IN);
    int o   = rem / C_IN;
    int c   = rem % C_IN;
    wp[idx] = (bf16)w[o * (C_IN * KTAPS) + c * KTAPS + tap];
}

// ---------------------------------------------------------------------------
// Implicit-GEMM 3x3 conv (pad=1, stride=1), bf16 WMMA + f32 accumulate.
// Block = 256 threads (8 wave32). Block tile = 128 och x (4 h-rows x 16 w).
// Per tap: TDM DMAs the A panel Wp[tap] (128x64 bf16) into padded LDS while
// all threads stage B^T (4 rows x 64 ch x 16 px, f32->bf16); then each wave
// runs 2 K-chunks x 4 N-tiles = 8 WMMAs, reusing each A fragment 4x.
// ---------------------------------------------------------------------------
__global__ __launch_bounds__(256) void conv3x3_wmma(const float* __restrict__ x,
                                                    const bf16* __restrict__ wp,
                                                    float* __restrict__ out) {
    __shared__ __align__(16) bf16 Alds[C_OUT * LDS_STR];     // [o][k0..63] 22.0 KB
    __shared__ __align__(16) bf16 Blds[NT * 16 * LDS_STR];   // [nt*16+p][k] 11.0 KB

    const int wb = blockIdx.x;            // 0..6   (w strip)
    const int hb = blockIdx.y;            // 0..27  (h strip of 4 rows)
    const int b  = blockIdx.z;            // 0..15
    const int w0 = wb * 16;
    const int h0 = hb * NT;

    const int tid   = threadIdx.x;
    const int lane  = tid & 31;
    const int wave  = tid >> 5;           // 0..7 -> M tile (16 out-channels)
    const int half  = lane >> 4;          // ISA A/B K-half selector
    const int l16   = lane & 15;
    const int o_row = wave * 16 + l16;

    // TDM writes Alds behind the compiler's back. Launder only the 32-bit LDS
    // *offset* through an empty asm (opaque provenance -> reads can't fold to
    // undef), then rebuild an addrspace(3) pointer so reads stay ds_load_b128.
    const unsigned lds_off = (unsigned)(uintptr_t)&Alds[0];  // low 32b = LDS offset
    unsigned a_off = lds_off;
    asm volatile("" : "+s"(a_off));
    lds_cbf16* Ab = (lds_cbf16*)(uintptr_t)a_off;

    v8f acc[NT] = {};

    const float* xb = x + (size_t)b * C_IN * HW * HW;

    for (int tap = 0; tap < KTAPS; ++tap) {
        const int dh = tap / 3 - 1;
        const int dw = tap % 3 - 1;

        __syncthreads();   // previous tap's LDS reads complete before overwrite

        // ---- TDM: DMA Wp[tap] (128 rows x 64 bf16, row stride 64) -> Alds,
        //      padding 12 dwords after every 32 dwords => LDS row stride 88.
        if (wave == 0) {
            unsigned long long ga = (unsigned long long)(uintptr_t)wp
                                  + (unsigned long long)tap * (C_OUT * C_IN * 2);
            // D# group0: count=1 | lds_addr | global_addr[56:0] | type=2
            u32x4 g0 = { 1u, lds_off, (unsigned)ga,
                         ((unsigned)((ga >> 32) & 0x01FFFFFFu)) | 0x80000000u };
            // D# group1: data_size=1(2B), pad_enable, pad_interval=4(32 dw),
            //            pad_amount=11(12 dw); tensor_dim0=64, tensor_dim1=128,
            //            tile_dim0=64, tile_dim1=128, tensor_dim0_stride=64
            i32x8 g1 = { 0x17110000, 64 << 16, 128 << 16, 64 << 16,
                         128, 64, 0, 0 };
            i32x4 z4 = { 0, 0, 0, 0 };
#if __clang_major__ >= 23
            i32x8 z8 = { 0, 0, 0, 0, 0, 0, 0, 0 };
            __builtin_amdgcn_tensor_load_to_lds(g0, g1, z4, z4, z8, 0);
#else
            __builtin_amdgcn_tensor_load_to_lds(g0, g1, z4, z4, 0);
#endif
        }

        // ---- stage B^T (branchless): x[b, kk, h0+nt+dh, w0+p+dw] -> Blds
        #pragma unroll
        for (int i = 0; i < 16; ++i) {
            int e  = tid + i * 256;         // 0..4095
            int p  = e & 15;                // pixel (coalesced across lanes)
            int kk = (e >> 4) & 63;         // channel = K row
            int nt = e >> 10;               // which h row
            int hy = h0 + nt + dh;
            int wx = w0 + p + dw;
            bool ok = ((unsigned)hy < (unsigned)HW) & ((unsigned)wx < (unsigned)HW);
            int hyc = ok ? hy : 0;
            int wxc = ok ? wx : 0;
            float v = xb[(size_t)kk * (HW * HW) + hyc * HW + wxc];
            Blds[(nt * 16 + p) * LDS_STR + kk] = (bf16)(ok ? v : 0.0f);
        }

        if (wave == 0) __builtin_amdgcn_s_wait_tensorcnt(0);  // A tile landed
        __syncthreads();                                      // publish A and B

        // ---- 2 K-chunks x 4 N-tiles, A fragment reused 4x per chunk
        #pragma unroll
        for (int cb = 0; cb < 2; ++cb) {
            const int c0 = cb * 32;
            // A lane: M=lane%16; elems[0..7]=k half*8.., elems[8..15]=k 16+half*8..
            v8bf a0 = *(lds_cv8bf*)(Ab + o_row * LDS_STR + c0 + half * 8);
            v8bf a1 = *(lds_cv8bf*)(Ab + o_row * LDS_STR + c0 + 16 + half * 8);
            v16bf af = __builtin_shufflevector(a0, a1, 0,1,2,3,4,5,6,7,
                                                        8,9,10,11,12,13,14,15);
            #pragma unroll
            for (int nt = 0; nt < NT; ++nt) {
                // B lane: N=lane%16; elems[0..7]=k half*8.., elems[8..15]=k 16+half*8..
                const bf16* bp = Blds + (nt * 16 + l16) * LDS_STR + c0 + half * 8;
                v8bf b0 = *(const v8bf*)(bp);
                v8bf b1 = *(const v8bf*)(bp + 16);
                v16bf bfr = __builtin_shufflevector(b0, b1, 0,1,2,3,4,5,6,7,
                                                             8,9,10,11,12,13,14,15);
                acc[nt] = __builtin_amdgcn_wmma_f32_16x16x32_bf16(
                              false, af, false, bfr, (short)0, acc[nt],
                              false, false);
            }
        }
    }

    // ---- store: C/D layout: VGPR r, lanes<16 -> M=r, lanes>=16 -> M=8+r
    #pragma unroll
    for (int nt = 0; nt < NT; ++nt) {
        const int h = h0 + nt;
        float* ob = out + (((size_t)b * C_OUT + wave * 16 + half * 8) * HW + h) * HW
                        + w0 + l16;
        #pragma unroll
        for (int r = 0; r < 8; ++r) {
            ob[(size_t)r * HW * HW] = acc[nt][r];   // 64B contiguous per 16 lanes
        }
    }
}

// ---------------------------------------------------------------------------
extern "C" void kernel_launch(void* const* d_in, const int* in_sizes, int n_in,
                              void* d_out, int out_size, void* d_ws, size_t ws_size,
                              hipStream_t stream) {
    const float* x = (const float*)d_in[0];   // (16, 64, 112, 112) f32
    const float* w = (const float*)d_in[1];   // (128, 64, 9) f32
    float* out = (float*)d_out;               // (16, 128, 112, 112) f32
    bf16* wp = (bf16*)d_ws;                   // 9*128*64 bf16 = 147456 B scratch

    pack_weights<<<dim3((KTAPS * C_OUT * C_IN) / 256), dim3(256), 0, stream>>>(w, wp);

    dim3 grid(HW / 16, HW / NT, 16);          // (7, 28, 16) = 3136 blocks
    conv3x3_wmma<<<grid, dim3(256), 0, stream>>>(x, wp, out);
}
